// state_transition_network_64295660421445
// MI455X (gfx1250) — matrix operand order repaired and checked by reading, not compiled, not run on hardware
//
#include <hip/hip_runtime.h>
#include <hip/hip_bf16.h>

// GRU-like scan:  gi = input@W^T + b (hoisted, bf16 WMMA)  then sequential
// h_t = tanh(gi_n + sigmoid(gi_r + h@W_hr^T + b_hr) * (h@W_hn^T + b_hn))
//
// Phase 1: big GEMM -> gi_r (ws) and gi_n (written into d_out, consumed and
//          overwritten in place by phase 2).
// Phase 2: 24 persistent workgroups, weights resident in LDS as pre-swizzled
//          WMMA B-fragments for all 1024 steps, async global->LDS staging of
//          the hidden state (global_load_async_to_lds_b128), global arrive-
//          counter barrier per step, double-buffered bf16 hidden state.

#define H   768
#define TT  1024
#define BB  32
#define NWG 24          // persistent workgroups in the scan (768 cols / 32)

typedef __attribute__((ext_vector_type(8)))  __bf16 v8bf;
typedef __attribute__((ext_vector_type(16))) __bf16 v16bf;
typedef __attribute__((ext_vector_type(8)))  float  v8f;

static __device__ inline unsigned int f2bf_bits(float f) {
    union { float f; unsigned int u; } v; v.f = f;
    unsigned int u = v.u;
    u += 0x7FFFu + ((u >> 16) & 1u);            // round to nearest even
    return u >> 16;
}
static __device__ inline unsigned short f2bf(float f) {
    return (unsigned short)f2bf_bits(f);
}
// two f32 -> packed bf16x2 (low = f0, high = f1)
static __device__ inline unsigned int pack2(float f0, float f1) {
    return f2bf_bits(f0) | (f2bf_bits(f1) << 16);
}
// eight consecutive f32 -> one uint4 of packed bf16
static __device__ inline uint4 pack8(const float* s) {
    uint4 r;
    r.x = pack2(s[0], s[1]); r.y = pack2(s[2], s[3]);
    r.z = pack2(s[4], s[5]); r.w = pack2(s[6], s[7]);
    return r;
}

// Build a 16-element bf16 fragment from two 16-byte LDS runs.
static __device__ inline v16bf ld_frag(const unsigned short* p0, const unsigned short* p1) {
    v8bf a = *(const v8bf*)p0;
    v8bf b = *(const v8bf*)p1;
    v16bf r;
#pragma unroll
    for (int i = 0; i < 8; ++i) { r[i] = a[i]; r[i + 8] = b[i]; }
    return r;
}

static __device__ inline v8f zero_v8f() {
    v8f z;
#pragma unroll
    for (int i = 0; i < 8; ++i) z[i] = 0.0f;
    return z;
}

// ---------------------------------------------------------------------------
// Init: h0 (f32) -> bf16 ping buffer, zero the barrier counters.
// ---------------------------------------------------------------------------
__global__ void gru_init_kernel(const float* __restrict__ h0,
                                unsigned short* __restrict__ hbuf,
                                unsigned int* __restrict__ cnt) {
    int i = blockIdx.x * blockDim.x + threadIdx.x;
    if (i < BB * H) hbuf[i] = f2bf(h0[i]);
    if (i < TT)     cnt[i]  = 0u;
}

// ---------------------------------------------------------------------------
// Phase 1: gi[m, n1536] = input[m, :] @ W_ih[rowmap(n), :]^T + b_ih[rowmap(n)]
// m = b*T + t (32768 rows).  n<768 -> gi_r (W row n), n>=768 -> gi_n (W row 2H+n-768).
// Tile: 128M x 64N per workgroup, K chunks of 32, 8 waves each doing 32x32.
// ---------------------------------------------------------------------------
__global__ __launch_bounds__(256) void gru_gemm_gi_kernel(
    const float* __restrict__ input,   // [B*T][H]
    const float* __restrict__ W_ih,    // [3H][H]
    const float* __restrict__ b_ih,    // [3H]
    float* __restrict__ gi_r,          // [B*T][H]  (ws)
    float* __restrict__ gi_n)          // [B*T][H]  (d_out)
{
    __shared__ unsigned short As[128 * 32];   // [m][k] bf16
    __shared__ unsigned short Bs[64 * 32];    // [n][k] bf16

    const int tid  = threadIdx.x;
    const int lane = tid & 31;
    const int w    = tid >> 5;
    const int wm   = w & 3;        // 4 M-subtiles of 32
    const int wn   = w >> 2;       // 2 N-subtiles of 32
    const int m0   = blockIdx.x * 128;
    const int n0   = blockIdx.y * 64;   // in [0, 1536)
    const int lm   = lane & 15;
    const int lh   = lane >> 4;

    v8f acc[2][2];
#pragma unroll
    for (int mi = 0; mi < 2; ++mi)
#pragma unroll
        for (int ni = 0; ni < 2; ++ni) acc[mi][ni] = zero_v8f();

    for (int kc = 0; kc < H / 32; ++kc) {
        __syncthreads();
        {   // stage A: 128x32, f32 -> bf16, packed uint4 stores
            int row = tid >> 1;
            int kq  = (tid & 1) * 16;
            const float* src = input + (size_t)(m0 + row) * H + kc * 32 + kq;
            float buf[16];
#pragma unroll
            for (int i = 0; i < 16; ++i) buf[i] = src[i];
            uint4* dst = (uint4*)&As[row * 32 + kq];
            dst[0] = pack8(buf);
            dst[1] = pack8(buf + 8);
        }
        {   // stage B: 64x32 from W_ih rows, f32 -> bf16, packed uint4 store
            int n  = tid >> 2;
            int kq = (tid & 3) * 8;
            int ng = n0 + n;
            int wrow = (ng < H) ? ng : (ng + H);     // gate r: n ; gate n: 2H+n
            const float* src = W_ih + (size_t)wrow * H + kc * 32 + kq;
            float buf[8];
#pragma unroll
            for (int i = 0; i < 8; ++i) buf[i] = src[i];
            *(uint4*)&Bs[n * 32 + kq] = pack8(buf);
        }
        __syncthreads();

#pragma unroll
        for (int mi = 0; mi < 2; ++mi) {
            int row = wm * 32 + mi * 16 + lm;
            const unsigned short* pa = &As[row * 32 + lh * 8];
            v16bf a = ld_frag(pa, pa + 16);
#pragma unroll
            for (int ni = 0; ni < 2; ++ni) {
                int n = wn * 32 + ni * 16 + lm;
                const unsigned short* pb = &Bs[n * 32 + lh * 8];
                v16bf b = ld_frag(pb, pb + 16);
                acc[mi][ni] = __builtin_amdgcn_wmma_f32_16x16x32_bf16(
                    false, a, false, b, (short)0, acc[mi][ni], false, false);
            }
        }
    }

#pragma unroll
    for (int mi = 0; mi < 2; ++mi)
#pragma unroll
        for (int ni = 0; ni < 2; ++ni) {
            int nglob = n0 + wn * 32 + ni * 16 + lm;
            int wrow  = (nglob < H) ? nglob : (nglob + H);
            float bias = b_ih[wrow];
            float* dst = (nglob < H) ? gi_r : gi_n;
            int ncol   = (nglob < H) ? nglob : (nglob - H);
#pragma unroll
            for (int v = 0; v < 8; ++v) {
                int m = m0 + wm * 32 + mi * 16 + v + lh * 8;
                dst[(size_t)m * H + ncol] = acc[mi][ni][v] + bias;
            }
        }
}

// ---------------------------------------------------------------------------
// Phase 2: persistent scan. 24 WGs x 32 columns. 8 waves = {mt}x{nt}x{gate}.
// LDS: hS 48KB (bf16 hidden state, at dynamic-LDS offset 0 -> raw LDS byte
// addresses usable by async loads), Bf 96KB (pre-swizzled B-fragments held
// for all 1024 steps), xch 4KB (r-gate accumulators), bias 256B.
// ---------------------------------------------------------------------------
__global__ __launch_bounds__(256) void gru_scan_kernel(
    const float* __restrict__ W_hh,          // [3H][H]
    const float* __restrict__ b_hh,          // [3H]
    const float* __restrict__ gi_r,          // [B*T][H]
    float* __restrict__ out,                 // d_out: gi_n in, h_t out (in place)
    unsigned short* __restrict__ hbuf,       // [2][B][H] bf16 ping-pong
    unsigned int* __restrict__ cnt)          // [T] arrive counters (zeroed)
{
    extern __shared__ unsigned char smem[];
    unsigned short* hS    = (unsigned short*)smem;           // BB*H (offset 0)
    unsigned short* Bf    = hS + BB * H;                     // 96 frags * 512
    float*          xch   = (float*)(Bf + 96 * 512);         // 4*256
    float*          biasS = xch + 4 * 256;                   // 64

    const int tid  = threadIdx.x;
    const int lane = tid & 31;
    const int w    = tid >> 5;
    const int mt   = w & 1;          // batch half (16 rows)
    const int nt   = (w >> 1) & 1;   // column half (16 cols)
    const int gate = w >> 2;         // 0 = r, 1 = n
    const int n0   = blockIdx.x * 32;
    const int lm   = lane & 15;
    const int lh   = lane >> 4;

    // Pre-swizzle W_hh slice into B fragments: frag f = (gate*2+nt)*24 + kc,
    // lane fl holds 16 consecutive K at k = kc*32 + (fl>=16)*16, col = nt*16+(fl&15).
#pragma unroll
    for (int i = 0; i < 12; ++i) {
        int idx = tid + 256 * i;          // 0..3071 (frag,lane) pairs
        int f   = idx >> 5;
        int fl  = idx & 31;
        int g   = f / 48;
        int rem = f % 48;
        int fnt = rem / 24;
        int fkc = rem % 24;
        int k   = fkc * 32 + (fl >> 4) * 16;
        int nl  = fnt * 16 + (fl & 15);
        int wrow = (g == 0) ? (n0 + nl) : (2 * H + n0 + nl);
        const float* src = W_hh + (size_t)wrow * H + k;
        float buf[16];
#pragma unroll
        for (int e = 0; e < 16; ++e) buf[e] = src[e];
        uint4* dst = (uint4*)&Bf[(size_t)f * 512 + fl * 16];
        dst[0] = pack8(buf);
        dst[1] = pack8(buf + 8);
    }
    if (tid < 32)       biasS[tid] = b_hh[n0 + tid];                 // b_hr
    else if (tid < 64)  biasS[tid] = b_hh[2 * H + n0 + (tid - 32)];  // b_hn

    const unsigned long long hbase = (unsigned long long)hbuf;

    for (int t = 0; t < TT; ++t) {
        __syncthreads();
        {   // async-stage hidden state (48KB bf16) global -> LDS, ASYNCcnt path
            unsigned lds0 = (unsigned)(tid * 16);                    // hS at LDS offset 0
            unsigned gof0 = (unsigned)((t & 1) * (BB * H * 2) + tid * 16);
#pragma unroll
            for (int i = 0; i < 12; ++i) {
                unsigned lds = lds0 + (unsigned)(i * 4096);
                unsigned gof = gof0 + (unsigned)(i * 4096);
                asm volatile("global_load_async_to_lds_b128 %0, %1, %2"
                             :: "v"(lds), "v"(gof), "s"(hbase)
                             : "memory");
            }
            asm volatile("s_wait_asynccnt 0x0" ::: "memory");
        }
        __syncthreads();

        // a = h @ W^T slice for this wave's (mt, nt, gate)
        v8f acc = zero_v8f();
        const unsigned short* abase = &hS[(mt * 16 + lm) * H + lh * 16];
        const unsigned short* bbase = &Bf[((size_t)(gate * 2 + nt) * 24) * 512 + lane * 16];
#pragma unroll
        for (int kc = 0; kc < H / 32; ++kc) {
            v16bf a = ld_frag(abase + kc * 32,  abase + kc * 32 + 8);
            v16bf b = ld_frag(bbase + kc * 512, bbase + kc * 512 + 8);
            acc = __builtin_amdgcn_wmma_f32_16x16x32_bf16(
                false, a, false, b, (short)0, acc, false, false);
        }

        if (gate == 0) {   // publish r-gate pre-activation
            float* x = &xch[(mt * 2 + nt) * 256];
#pragma unroll
            for (int v = 0; v < 8; ++v) x[v * 32 + lane] = acc[v];
        }
        __syncthreads();

        if (gate == 1) {   // combine: r = sigmoid(...), h = tanh(...)
            const float* x = &xch[(mt * 2 + nt) * 256];
            float bhr = biasS[nt * 16 + lm];
            float bhn = biasS[32 + nt * 16 + lm];
#pragma unroll
            for (int v = 0; v < 8; ++v) {
                int b  = mt * 16 + v + lh * 8;            // batch index
                int ng = n0 + nt * 16 + lm;               // global column
                size_t off = (size_t)b * (TT * H) + (size_t)t * H + ng;
                float a_r = x[v * 32 + lane];
                float r   = 1.0f / (1.0f + __expf(-(gi_r[off] + a_r + bhr)));
                float hn  = tanhf(out[off] + r * (acc[v] + bhn));   // out[off] = gi_n
                out[off] = hn;                                       // overwrite in place
                hbuf[(size_t)((t + 1) & 1) * BB * H + (size_t)b * H + ng] = f2bf(hn);
            }
        } else if (t + 1 < TT) {   // idle r-waves prefetch next step's gi rows
            int b = mt * 16 + lm;
            size_t off = (size_t)b * (TT * H) + (size_t)(t + 1) * H + n0;
            __builtin_prefetch(&gi_r[off], 0, 1);
            __builtin_prefetch(&out[off], 0, 1);
        }

        // device-wide step barrier (release writes, acquire reads)
        __threadfence();
        __syncthreads();
        if (tid == 0) {
            __hip_atomic_fetch_add(&cnt[t], 1u, __ATOMIC_RELEASE, __HIP_MEMORY_SCOPE_AGENT);
            while (__hip_atomic_load(&cnt[t], __ATOMIC_ACQUIRE, __HIP_MEMORY_SCOPE_AGENT)
                   < (unsigned)gridDim.x) {
                __builtin_amdgcn_s_sleep(2);
            }
        }
        __syncthreads();
        __threadfence();
    }
}

// ---------------------------------------------------------------------------
extern "C" void kernel_launch(void* const* d_in, const int* in_sizes, int n_in,
                              void* d_out, int out_size, void* d_ws, size_t ws_size,
                              hipStream_t stream) {
    const float* input  = (const float*)d_in[0];   // (B,T,H)
    const float* h0     = (const float*)d_in[1];   // (1,B,H)
    const float* W_ih   = (const float*)d_in[2];   // (3H,H)
    const float* W_hh   = (const float*)d_in[3];   // (3H,H)
    const float* b_ih   = (const float*)d_in[4];   // (3H)
    const float* b_hh   = (const float*)d_in[5];   // (3H)
    float*       out    = (float*)d_out;           // (B,T,H)

    // workspace layout
    unsigned char* ws = (unsigned char*)d_ws;
    const size_t gi_r_bytes = (size_t)BB * TT * H * sizeof(float);   // 96 MB
    float*          gi_r = (float*)ws;
    unsigned short* hbuf = (unsigned short*)(ws + gi_r_bytes);       // 2*B*H bf16
    unsigned int*   cnt  = (unsigned int*)(ws + gi_r_bytes + 2 * BB * H * sizeof(unsigned short));

    // init hidden state + barrier counters
    gru_init_kernel<<<(BB * H + 255) / 256, 256, 0, stream>>>(h0, hbuf, cnt);

    // phase 1: gi_r -> ws, gi_n -> d_out
    dim3 g1(BB * TT / 128, (2 * H) / 64);
    gru_gemm_gi_kernel<<<g1, 256, 0, stream>>>(input, W_ih, b_ih, gi_r, out);

    // phase 2: persistent scan
    const int smem = (BB * H) * 2 + 96 * 512 * 2 + 4 * 256 * 4 + 64 * 4;  // ~151.8 KB
    (void)hipFuncSetAttribute((const void*)gru_scan_kernel,
                              hipFuncAttributeMaxDynamicSharedMemorySize, smem);
    gru_scan_kernel<<<NWG, 256, smem, stream>>>(W_hh, b_hh, gi_r, out, hbuf, cnt);
}
